// MambaMixer_17136919511394
// MI455X (gfx1250) — matrix operand order repaired and checked by reading, not compiled, not run on hardware
//
#include <hip/hip_runtime.h>

// Problem constants (from the reference): H=2048, I=4096, N=16, R=128, K=4, B=1, L=2048
#define HH   2048
#define II   4096
#define SN   16
#define RR   128
#define KK   4
#define LL   2048
#define TWOI 8192
#define GN   160   // R + 2N

typedef __attribute__((ext_vector_type(16))) __bf16 v16bf;
typedef __attribute__((ext_vector_type(8)))  __bf16 v8bf;
typedef __attribute__((ext_vector_type(8)))  float  v8f;

// ---------------------------------------------------------------------------
// Weight transpose + fp32 -> bf16 : out[n*K + k] = (bf16) in[k*N + n]
// ---------------------------------------------------------------------------
__global__ __launch_bounds__(256) void convT_kernel(__bf16* __restrict__ out,
                                                    const float* __restrict__ in,
                                                    int K, int N) {
  long idx = (long)blockIdx.x * blockDim.x + threadIdx.x;
  long total = (long)K * N;
  if (idx >= total) return;
  int n = (int)(idx / K);
  int k = (int)(idx % K);
  out[(long)n * K + k] = (__bf16)in[(long)k * N + n];
}

// ---------------------------------------------------------------------------
// bf16 WMMA GEMM:  C[M,N] (fp32) = A[M,K] (fp32, row-major, lda)
//                                x W[K,N] given pre-transposed bf16 WT[N][K]
// Block: 256 threads = 8 waves in a 4(M) x 2(N) grid. Block tile 128 x 128.
// Wave tile 32(M) x 64(N): 8 WMMAs per 32-K step sharing 2 A-frags + 4 B-frags
// (12 ds_load_b128 : 8 v_wmma). LDS tiles are double-buffered: next K-step's
// global_load_b128s issue before the WMMAs; one barrier per K-step.
// Requires: M % 128 == 0, K % 32 == 0 (true at all call sites). N tail handled.
// ---------------------------------------------------------------------------
__global__ __launch_bounds__(256) void gemm_bf16(const float* __restrict__ A, int lda,
                                                 const __bf16* __restrict__ WT,
                                                 float* __restrict__ C, int ldc,
                                                 int M, int N, int K) {
  __shared__ __bf16 lA[2][128 * 40];   // 128 rows x 32 K, padded stride 40
  __shared__ __bf16 lB[2][128 * 32];   // 128 N-rows x 32 K, contiguous per row

  const int tid  = threadIdx.x;
  const int wave = tid >> 5;
  const int lane = tid & 31;
  const int half = lane >> 4;       // 0 or 1
  const int l16  = lane & 15;
  const int wm   = wave >> 1;       // 0..3 : M position of wave
  const int wn   = wave & 1;        // 0..1 : N position of wave
  const int mbase = blockIdx.y * 128;
  const int nbase = blockIdx.x * 128;

  v8f acc[2][4];
#pragma unroll
  for (int a = 0; a < 2; ++a)
#pragma unroll
    for (int j = 0; j < 4; ++j)
#pragma unroll
      for (int r = 0; r < 8; ++r) acc[a][j][r] = 0.0f;

  // staging assignment: thread copies 16 contiguous elements of one tile row
  const int trow = tid >> 1;          // 0..127
  const int tcol = (tid & 1) * 16;    // 0 or 16

  const float*  aptr = A + (long)(mbase + trow) * lda + tcol;
  const int     ngRow = nbase + trow;
  const __bf16* bptr = (ngRow < N) ? (WT + (long)ngRow * K + tcol) : (const __bf16*)0;

  float4 fa0, fa1, fa2, fa3;
  uint4  fb0, fb1;

  auto gload = [&](int kb) {
    const float4* ap4 = (const float4*)(aptr + kb);
    fa0 = ap4[0]; fa1 = ap4[1]; fa2 = ap4[2]; fa3 = ap4[3];
    if (bptr) {
      const uint4* bp4 = (const uint4*)(bptr + kb);
      fb0 = bp4[0]; fb1 = bp4[1];
    } else {
      fb0 = make_uint4(0u, 0u, 0u, 0u);
      fb1 = make_uint4(0u, 0u, 0u, 0u);
    }
  };
  auto sstore = [&](int buf) {
    v8bf b0, b1;
    b0[0] = (__bf16)fa0.x; b0[1] = (__bf16)fa0.y; b0[2] = (__bf16)fa0.z; b0[3] = (__bf16)fa0.w;
    b0[4] = (__bf16)fa1.x; b0[5] = (__bf16)fa1.y; b0[6] = (__bf16)fa1.z; b0[7] = (__bf16)fa1.w;
    b1[0] = (__bf16)fa2.x; b1[1] = (__bf16)fa2.y; b1[2] = (__bf16)fa2.z; b1[3] = (__bf16)fa2.w;
    b1[4] = (__bf16)fa3.x; b1[5] = (__bf16)fa3.y; b1[6] = (__bf16)fa3.z; b1[7] = (__bf16)fa3.w;
    *(v8bf*)&lA[buf][trow * 40 + tcol]     = b0;
    *(v8bf*)&lA[buf][trow * 40 + tcol + 8] = b1;
    uint4* dst = (uint4*)&lB[buf][trow * 32 + tcol];
    dst[0] = fb0;
    dst[1] = fb1;
  };

  gload(0);
  sstore(0);
  __syncthreads();

  const int nk = K >> 5;
  for (int t = 0; t < nk; ++t) {
    const int cur = t & 1;
    const bool more = (t + 1) < nk;
    if (more) gload((t + 1) << 5);   // overlap next tile's global loads with WMMAs

    // A fragments in contiguous k order (not the ISA table order):
    // lane<16 : row M, K 0..15;  lane>=16 : K 16..31 (contiguous, matches B)
    v16bf af[2];
#pragma unroll
    for (int a = 0; a < 2; ++a) {
      const __bf16* ar = &lA[cur][(wm * 32 + a * 16 + l16) * 40];
      v8bf alo = *(const v8bf*)(ar + half * 16);
      v8bf ahi = *(const v8bf*)(ar + half * 16 + 8);
      af[a] = __builtin_shufflevector(alo, ahi,
          0, 1, 2, 3, 4, 5, 6, 7, 8, 9, 10, 11, 12, 13, 14, 15);
    }
#pragma unroll
    for (int j = 0; j < 4; ++j) {
      // B fragment: lane<16 -> col N, K 0..15 ; lane>=16 -> K 16..31 (contiguous)
      v16bf bfrag = *(const v16bf*)&lB[cur][(wn * 64 + j * 16 + l16) * 32 + half * 16];
      acc[0][j] = __builtin_amdgcn_wmma_f32_16x16x32_bf16(
          false, af[0], false, bfrag, (short)0, acc[0][j], false, false);
      acc[1][j] = __builtin_amdgcn_wmma_f32_16x16x32_bf16(
          false, af[1], false, bfrag, (short)0, acc[1][j], false, false);
    }
    if (more) sstore(1 - cur);       // write next buffer; others may still read cur
    __syncthreads();
  }

  // ---- store: VGPR r holds M = r + 8*half, N = l16 ----
#pragma unroll
  for (int a = 0; a < 2; ++a) {
    const int rowb = mbase + wm * 32 + a * 16 + half * 8;
#pragma unroll
    for (int j = 0; j < 4; ++j) {
      int col = nbase + wn * 64 + j * 16 + l16;
      if (col < N) {
#pragma unroll
        for (int r = 0; r < 8; ++r) {
          C[(long)(rowb + r) * ldc + col] = acc[a][j][r];
        }
      }
    }
  }
}

// ---------------------------------------------------------------------------
// Depthwise causal conv1d (K=4, left pad 3) + bias + SiLU.
// h channel i lives in proj[:, i] (first I of the 2I projection columns).
// Output hconv[l*I + i].
// ---------------------------------------------------------------------------
__global__ __launch_bounds__(256) void conv_silu_kernel(const float* __restrict__ proj,
                                                        const float* __restrict__ conv_w,
                                                        const float* __restrict__ conv_b,
                                                        float* __restrict__ hconv) {
  int idx = blockIdx.x * blockDim.x + threadIdx.x;   // over L*I = 8M
  int l = idx / II;
  int i = idx - l * II;
  float w0 = conv_w[i * KK + 0], w1 = conv_w[i * KK + 1];
  float w2 = conv_w[i * KK + 2], w3 = conv_w[i * KK + 3];
  const float* pc = proj + (long)l * TWOI + i;
  float acc = conv_b[i] + w3 * pc[0];
  if (l >= 1) acc += w2 * pc[-1 * TWOI];
  if (l >= 2) acc += w1 * pc[-2 * TWOI];
  if (l >= 3) acc += w0 * pc[-3 * TWOI];
  hconv[(long)l * II + i] = acc / (1.0f + __expf(-acc));   // SiLU
}

// ---------------------------------------------------------------------------
// Selective scan + fused epilogue.
// Each half-wave (16 lanes) owns channel i's 16 SSM states: lane n holds s[i,n].
// dt (softplus w/ bias), dA = exp(A*dt), dBu = dt*B*h computed on the fly —
// avoids materializing the 512MB [B,I,L,N] tensors. Only the per-step FMA
// s = dA*s + dBu is on the serial chain. y reduced with 4x shfl_xor, then
// fused with (y + h*D) * silu(gate) and written for the output GEMM.
// ---------------------------------------------------------------------------
__global__ __launch_bounds__(256) void scan_kernel(const float* __restrict__ dtraw,
                                                   const float* __restrict__ ssm,
                                                   const float* __restrict__ hconv,
                                                   const float* __restrict__ proj,
                                                   const float* __restrict__ A_log,
                                                   const float* __restrict__ b_dt,
                                                   const float* __restrict__ Dp,
                                                   float* __restrict__ ycomb) {
  const int tid  = threadIdx.x;
  const int wave = tid >> 5;
  const int lane = tid & 31;
  const int half = lane >> 4;
  const int n    = lane & 15;
  const int i    = blockIdx.x * 16 + wave * 2 + half;   // channel

  const float Ac  = -__expf(A_log[i * SN + n]);   // A = -exp(A_log)
  const float bdt = b_dt[i];
  const float Dv  = Dp[i];
  const bool writer = (n == 0);

  float s = 0.0f;
  for (int l = 0; l < LL; ++l) {
    float Bv  = ssm[(long)l * GN + RR + n];
    float Cv  = ssm[(long)l * GN + RR + SN + n];
    float dtv = dtraw[(long)l * II + i] + bdt;
    float h   = hconv[(long)l * II + i];
    if (l + 8 < LL) {
      __builtin_prefetch(&dtraw[(long)(l + 8) * II + i], 0, 0);
      __builtin_prefetch(&hconv[(long)(l + 8) * II + i], 0, 0);
    }
    float sp = (dtv > 20.0f) ? dtv : log1pf(__expf(dtv));   // softplus
    float dA = __expf(Ac * sp);
    s = dA * s + sp * Bv * h;

    float part = s * Cv;
    part += __shfl_xor(part, 1);
    part += __shfl_xor(part, 2);
    part += __shfl_xor(part, 4);
    part += __shfl_xor(part, 8);

    if (writer) {
      float g  = proj[(long)l * TWOI + II + i];            // gate
      float sg = g / (1.0f + __expf(-g));                  // silu(gate)
      ycomb[(long)l * II + i] = (part + h * Dv) * sg;
    }
  }
}

// ---------------------------------------------------------------------------
extern "C" void kernel_launch(void* const* d_in, const int* in_sizes, int n_in,
                              void* d_out, int out_size, void* d_ws, size_t ws_size,
                              hipStream_t stream) {
  const float* x      = (const float*)d_in[0];   // [L,H]
  const float* W_in   = (const float*)d_in[1];   // [H,2I]
  const float* conv_w = (const float*)d_in[2];   // [I,1,K]
  const float* conv_b = (const float*)d_in[3];   // [I]
  const float* W_x    = (const float*)d_in[4];   // [I,R+2N]
  const float* W_dt   = (const float*)d_in[5];   // [R,I]
  const float* b_dt   = (const float*)d_in[6];   // [I]
  const float* W_out  = (const float*)d_in[7];   // [I,H]
  const float* A_log  = (const float*)d_in[8];   // [I,N]
  const float* Dp     = (const float*)d_in[9];   // [I]
  float* out = (float*)d_out;                    // [L,H]
  (void)in_sizes; (void)n_in; (void)out_size; (void)ws_size;

  char* ws = (char*)d_ws;
  size_t off = 0;
  auto carve = [&](size_t bytes) -> char* {
    char* p = ws + off;
    off += (bytes + 255) & ~(size_t)255;
    return p;
  };
  __bf16* WinT  = (__bf16*)carve((size_t)TWOI * HH * 2);   //  32 MB  [2I][H]
  float*  proj  = (float*) carve((size_t)LL * TWOI * 4);   //  64 MB  [L][2I]
  float*  hconv = (float*) carve((size_t)LL * II * 4);     //  32 MB  [L][I]
  __bf16* WxT   = (__bf16*)carve((size_t)GN * II * 2);     // 1.3 MB  [160][I]
  float*  ssm   = (float*) carve((size_t)LL * GN * 4);     // 1.3 MB  [L][160]
  __bf16* WdtT  = (__bf16*)carve((size_t)II * RR * 2);     //   1 MB  [I][R]
  float*  dtraw = (float*) carve((size_t)LL * II * 4);     //  32 MB  [L][I]
  float*  ycomb = (float*) carve((size_t)LL * II * 4);     //  32 MB  [L][I]
  __bf16* WoutT = (__bf16*)carve((size_t)HH * II * 2);     //  16 MB  [H][I]

  // 1) transpose + convert all weights to bf16 [N][K]
  convT_kernel<<<((long)HH * TWOI + 255) / 256, 256, 0, stream>>>(WinT,  W_in,  HH, TWOI);
  convT_kernel<<<((long)II * GN + 255) / 256, 256, 0, stream>>>(WxT,   W_x,   II, GN);
  convT_kernel<<<((long)RR * II + 255) / 256, 256, 0, stream>>>(WdtT,  W_dt,  RR, II);
  convT_kernel<<<((long)II * HH + 255) / 256, 256, 0, stream>>>(WoutT, W_out, II, HH);

  dim3 blk(256);
  // 2) in-projection: proj[L,2I] = x[L,H] @ W_in
  gemm_bf16<<<dim3(TWOI / 128, LL / 128), blk, 0, stream>>>(x, HH, WinT, proj, TWOI,
                                                            LL, TWOI, HH);
  // 3) causal depthwise conv + SiLU -> hconv[L,I]
  conv_silu_kernel<<<(LL * II) / 256, 256, 0, stream>>>(proj, conv_w, conv_b, hconv);
  // 4) x-projection: ssm[L,160] = hconv @ W_x     (N tail handled)
  gemm_bf16<<<dim3((GN + 127) / 128, LL / 128), blk, 0, stream>>>(hconv, II, WxT, ssm, GN,
                                                                  LL, GN, II);
  // 5) dt-projection: dtraw[L,I] = ssm[:, :R] @ W_dt   (lda=160, K=128)
  gemm_bf16<<<dim3(II / 128, LL / 128), blk, 0, stream>>>(ssm, GN, WdtT, dtraw, II,
                                                          LL, II, RR);
  // 6) selective scan + fused (y + h*D)*silu(gate) -> ycomb[L,I]
  scan_kernel<<<II / 16, 256, 0, stream>>>(dtraw, ssm, hconv, proj, A_log, b_dt, Dp, ycomb);
  // 7) out-projection: out[L,H] = ycomb @ W_out
  gemm_bf16<<<dim3(HH / 128, LL / 128), blk, 0, stream>>>(ycomb, II, WoutT, out, HH,
                                                          LL, HH, II);
}
